// SPF_LUT_DFC_10101763080281
// MI455X (gfx1250) — matrix operand order repaired and checked by reading, not compiled, not run on hardware
//
#include <hip/hip_runtime.h>

// SPF-LUT-DFC pipeline for MI455X (gfx1250, wave32).
//   - conv bank (4 rot x 4 ch per mode) via v_wmma_f32_16x16x32_f16 over a 5x5 im2col
//   - LUT tables staged into LDS with tensor_load_to_lds (TDM) + s_wait_tensorcnt
//   - WMMA A-operands precomputed into LDS (branch-free), fetched via ds_load_b128
//   - per-lane simplex interpolation, lane p / p+16 combined with __shfl_xor
// ref2index is arange(R1) => identity index, computed arithmetically.

typedef __attribute__((ext_vector_type(16))) _Float16 v16h;
typedef __attribute__((ext_vector_type(8)))  float    v8f;

#define HW     192
#define IMG_N  (HW * HW)
#define NPIX   (2 * IMG_N)

__device__ __forceinline__ int   spf_imin(int a, int b)   { return a < b ? a : b; }
__device__ __forceinline__ int   spf_imax(int a, int b)   { return a > b ? a : b; }
__device__ __forceinline__ float spf_clampf(float v, float lo, float hi) {
  return fminf(fmaxf(v, lo), hi);
}

// ---------------------------------------------------------------------------
// Tensor Data Mover: 1-D dword copy global -> LDS (D# per CDNA5 ISA ch.8).
// ---------------------------------------------------------------------------
#if __has_builtin(__builtin_amdgcn_tensor_load_to_lds)
#define SPF_USE_TDM 1
typedef __attribute__((ext_vector_type(4))) unsigned int tdm_u32x4;
typedef __attribute__((ext_vector_type(8))) int          tdm_i32x8;
typedef __attribute__((ext_vector_type(4))) int          tdm_i32x4;

__device__ __forceinline__ void tdm_copy_to_lds(void* lds_dst, const void* gsrc,
                                                unsigned ndwords) {
  unsigned long long ga = (unsigned long long)gsrc;
  tdm_u32x4 g0;
  g0.x = 1u;                                          // count=1 (valid), user mode
  g0.y = (unsigned)(unsigned long long)lds_dst;       // LDS byte address (low 32 bits)
  g0.z = (unsigned)ga;                                // global_addr[31:0]
  g0.w = (unsigned)((ga >> 32) & 0x1FFFFFFull) | (2u << 30); // addr[56:32] | type=2
  tdm_i32x8 g1;
  g1[0] = (int)(2u << 16);                            // data_size = 4 bytes
  g1[1] = (int)((ndwords & 0xFFFFu) << 16);           // tensor_dim0[15:0]
  g1[2] = (int)(((ndwords >> 16) & 0xFFFFu) | (1u << 16)); // tensor_dim0[31:16], tensor_dim1=1
  g1[3] = (int)((ndwords & 0xFFFFu) << 16);           // tile_dim0 = ndwords (<=65535)
  g1[4] = 0;                                          // tile_dim1=0 (unused) -> 1-D tile
  g1[5] = (int)ndwords;                               // tensor_dim0_stride
  g1[6] = 0;
  g1[7] = 0;
  tdm_i32x4 gz = {0, 0, 0, 0};
#if defined(__clang_major__) && (__clang_major__ >= 23)
  tdm_i32x8 gz8 = {0, 0, 0, 0, 0, 0, 0, 0};
  __builtin_amdgcn_tensor_load_to_lds(g0, g1, gz, gz, gz8, 0);
#else
  __builtin_amdgcn_tensor_load_to_lds(g0, g1, gz, gz, 0);
#endif
}
#endif

__device__ __forceinline__ void blk_copy_to_lds(float* dst, const float* src, int n) {
  for (int i = (int)threadIdx.x; i < n; i += (int)blockDim.x) dst[i] = src[i];
}

// ---------------------------------------------------------------------------
// Stable descending 4-sort (matches jnp.argsort(-f), stable on ties).
// ---------------------------------------------------------------------------
__device__ __forceinline__ void spf_sort4_desc(const float f[4], int ord[4]) {
  ord[0] = 0; ord[1] = 1; ord[2] = 2; ord[3] = 3;
#pragma unroll
  for (int i = 1; i < 4; ++i) {
    const int key = ord[i];
    const float fk = f[key];
    int j = i - 1;
#pragma unroll
    for (; j >= 0; --j) {
      if (f[ord[j]] < fk) ord[j + 1] = ord[j];
      else break;
    }
    ord[j + 1] = key;
  }
}

// ---------------------------------------------------------------------------
// One simplex-LUT term: v4 = 4 clamped conv outputs in [0,255].
// w1t: fine table (2125 x OC), w2t: coarse table (625 x OC).
// ---------------------------------------------------------------------------
template <int OC>
__device__ __forceinline__ void lut_term(const float v4[4],
                                         const float* __restrict__ w1t,
                                         const float* __restrict__ w2t,
                                         float* __restrict__ term) {
  int i1[4], i2[4];
  float f1[4], f2[4];
#pragma unroll
  for (int c = 0; c < 4; ++c) {
    const float v = v4[c];
    int a2 = (int)floorf(v * 0.015625f);           // /64
    a2 = spf_imin(spf_imax(a2, 0), 3);
    i2[c] = a2; f2[c] = v - (float)a2 * 64.f;
    int a1 = (int)floorf(v * 0.0625f);             // /16
    a1 = spf_imin(spf_imax(a1, 0), 15);
    i1[c] = a1; f1[c] = v - (float)a1 * 16.f;
  }
#pragma unroll
  for (int oc = 0; oc < OC; ++oc) term[oc] = 0.f;

  const int d1 = i1[1] - i1[0], d2 = i1[2] - i1[0], d3 = i1[3] - i1[0];
  const bool within = (d1 >= -1 && d1 <= 1) && (d2 >= -1 && d2 <= 1) &&
                      (d3 >= -1 && d3 <= 1);

  if (within) {
    int ord[4]; spf_sort4_desc(f1, ord);
    int rk[4];
#pragma unroll
    for (int k = 0; k < 4; ++k) rk[ord[k]] = k;
    const float s0 = f1[ord[0]], s1 = f1[ord[1]], s2 = f1[ord[2]], s3 = f1[ord[3]];
    const float wk[5] = {(16.f - s0) * 0.0625f, (s0 - s1) * 0.0625f,
                         (s1 - s2) * 0.0625f,   (s2 - s3) * 0.0625f,
                         s3 * 0.0625f};
#pragma unroll
    for (int k = 0; k < 5; ++k) {
      const int vx0 = spf_imin(i1[0] + (rk[0] < k ? 1 : 0), 16);
      const int vx1 = spf_imin(i1[1] + (rk[1] < k ? 1 : 0), 16);
      const int vx2 = spf_imin(i1[2] + (rk[2] < k ? 1 : 0), 16);
      const int vx3 = spf_imin(i1[3] + (rk[3] < k ? 1 : 0), 16);
      const int o0 = spf_imin(spf_imax(vx1 - vx0 + 2, 0), 4);
      const int o1 = spf_imin(spf_imax(vx2 - vx0 + 2, 0), 4);
      const int o2 = spf_imin(spf_imax(vx3 - vx0 + 2, 0), 4);
      const int row = ((vx0 * 5 + o0) * 5 + o1) * 5 + o2;  // ref2index identity
      const float* t = w1t + row * OC;
      const float wkk = wk[k];
#pragma unroll
      for (int oc = 0; oc < OC; ++oc) term[oc] += wkk * t[oc];
    }
  } else {
    int ord[4]; spf_sort4_desc(f2, ord);
    int rk[4];
#pragma unroll
    for (int k = 0; k < 4; ++k) rk[ord[k]] = k;
    const float s0 = f2[ord[0]], s1 = f2[ord[1]], s2 = f2[ord[2]], s3 = f2[ord[3]];
    const float wk[5] = {(64.f - s0) * 0.015625f, (s0 - s1) * 0.015625f,
                         (s1 - s2) * 0.015625f,   (s2 - s3) * 0.015625f,
                         s3 * 0.015625f};
#pragma unroll
    for (int k = 0; k < 5; ++k) {
      const int vx0 = spf_imin(i2[0] + (rk[0] < k ? 1 : 0), 4);
      const int vx1 = spf_imin(i2[1] + (rk[1] < k ? 1 : 0), 4);
      const int vx2 = spf_imin(i2[2] + (rk[2] < k ? 1 : 0), 4);
      const int vx3 = spf_imin(i2[3] + (rk[3] < k ? 1 : 0), 4);
      const int flat = ((vx0 * 5 + vx1) * 5 + vx2) * 5 + vx3;
      const float* t = w2t + flat * OC;
      const float wkk = wk[k];
#pragma unroll
      for (int oc = 0; oc < OC; ++oc) term[oc] += wkk * t[oc];
    }
  }
}

// ---------------------------------------------------------------------------
// Branch-free im2col column (clamped 5x5 neighborhood, taps K=0..24).
// B (32x16) lane layout: column = lane%16, lane half holds contiguous K range
// (lanes<16: K=0..15, lanes>=16: K=16..31; K>=25 is zero padding).
// (u,v) are unroll-time constants selected by `hi` -> no exec-mask branches.
// ---------------------------------------------------------------------------
__device__ __forceinline__ v16h build_B(const float* __restrict__ im, int y, int x,
                                        int hi) {
  v16h B;
#pragma unroll
  for (int h = 0; h < 16; ++h) {
    const int Klo = h;                       // lanes < 16
    const int Khi = 16 + h;                  // lanes >= 16
    const int u = (hi ? Khi / 5 : Klo / 5) - 2;
    const int v = (hi ? Khi % 5 : Klo % 5) - 2;
    const bool valid = hi ? (Khi < 25) : true;
    const int iy = spf_imin(spf_imax(y + u, 0), HW - 1);
    const int ix = spf_imin(spf_imax(x + v, 0), HW - 1);
    const float val = im[iy * HW + ix];      // unconditional, always in-bounds
    B[h] = (_Float16)(valid ? val : 0.f);    // v_cndmask, not a branch
  }
  return B;
}

// ---------------------------------------------------------------------------
// Cooperative fill of WMMA A-operands (rotated 3x3 kernels scattered into the
// 5x5 tap space) into LDS. One v16h per (combo, lane); identical for all waves.
// A (16x32) lane layout: row = lane%16; lanes<16 hold K 0..15,
// lanes>=16 hold K 16..31.
//   combo = mode (stage kernels)   -> sw stride 4*9 per combo
//   combo = m*4+c (stage6 kernel)  -> sw stride 4*9 per combo
// ---------------------------------------------------------------------------
__device__ __forceinline__ void fill_A_lds(_Float16* __restrict__ Ah,
                                           const float* __restrict__ sw,
                                           int ncombo) {
  const int total = ncombo * 32 * 16;
  for (int e = (int)threadIdx.x; e < total; e += 256) {
    const int combo = e >> 9;
    const int ln    = (e >> 4) & 31;
    const int h     = e & 15;
    const int hi_   = ln >> 4;
    const int row   = ln & 15;
    const int r     = row >> 2;
    const int ch    = row & 3;
    const int K = hi_ * 16 + h;
    float wv = 0.f;
    if (K < 25) {
      const int u = K / 5 - 2, v = K % 5 - 2;
      int di, dj;
      if (r == 0)      { di = u;  dj = v;  }
      else if (r == 1) { di = -v; dj = u;  }
      else if (r == 2) { di = -u; dj = -v; }
      else             { di = v;  dj = -u; }
      if (di >= 0 && di <= 2 && dj >= 0 && dj <= 2)
        wv = sw[(combo * 4 + ch) * 9 + di * 3 + dj];
    }
    Ah[e] = (_Float16)wv;
  }
}

// ---------------------------------------------------------------------------
// Stages 1-4: per-pixel sum over (m,r) of conv+simplex-LUT; OC = 2 or 1.
// Each wave owns 16 consecutive pixels in one row; one WMMA per mode covers
// all 4 rotations x 4 channels. LUT tables for the current mode live in LDS.
// ---------------------------------------------------------------------------
template <int OC>
__global__ void __launch_bounds__(256)
spf_stage_kernel(const float* __restrict__ img,   // (2,192,192) rounded ints
                 const float* __restrict__ sw,    // (3,4,3,3)
                 const float* __restrict__ w1,    // (3,2125,OC)
                 const float* __restrict__ w2,    // (3,625,OC)
                 const float* __restrict__ res,   // (3,OC)
                 float* __restrict__ out0, int out0_stride,
                 float* __restrict__ out1, int out1_stride) {
  __shared__ float w1s[2125 * OC];
  __shared__ float w2s[625 * OC];
  __shared__ v16h  Als[3 * 32];     // 3 modes x 32 lanes

  fill_A_lds((_Float16*)Als, sw, 3);

  const int lane = (int)threadIdx.x & 31;
  const int hi   = lane >> 4;
  const int wave = (int)blockIdx.x * 8 + ((int)threadIdx.x >> 5);
  const int p    = wave * 16 + (lane & 15);
  const int n    = p / IMG_N;
  const int rem  = p - n * IMG_N;
  const int y    = rem / HW;
  const int x    = rem - y * HW;

  const v16h B = build_B(img + n * IMG_N, y, x, hi);

  float accv[OC];
#pragma unroll
  for (int oc = 0; oc < OC; ++oc) accv[oc] = 0.f;

  for (int m = 0; m < 3; ++m) {
#if defined(SPF_USE_TDM)
    if (threadIdx.x < 32) {
      tdm_copy_to_lds(w1s, w1 + m * 2125 * OC, 2125u * OC);
      tdm_copy_to_lds(w2s, w2 + m * 625 * OC, 625u * OC);
      __builtin_amdgcn_s_wait_tensorcnt(0);
    }
#else
    blk_copy_to_lds(w1s, w1 + m * 2125 * OC, 2125 * OC);
    blk_copy_to_lds(w2s, w2 + m * 625 * OC, 625 * OC);
#endif
    __syncthreads();  // tables (and, on m==0, Als) visible to all waves

    const v16h A = Als[m * 32 + lane];   // ds_load_b128 x2

    v8f C = {};
    C = __builtin_amdgcn_wmma_f32_16x16x32_f16(false, A, false, B, (short)0, C,
                                               false, false);

    // lane holds 2 rotations x 4 channels for its pixel
#pragma unroll
    for (int g = 0; g < 2; ++g) {
      float v4[4];
#pragma unroll
      for (int c = 0; c < 4; ++c) v4[c] = spf_clampf(C[g * 4 + c], 0.f, 255.f);
      float term[OC];
      lut_term<OC>(v4, w1s, w2s, term);
#pragma unroll
      for (int oc = 0; oc < OC; ++oc) accv[oc] += term[oc] + res[m * OC + oc];
    }
    __syncthreads();  // before TDM overwrites LDS for next mode
  }

#pragma unroll
  for (int oc = 0; oc < OC; ++oc) accv[oc] += __shfl_xor(accv[oc], 16, 32);

  if (lane < 16) {
    const float q0 = accv[0] * (1.f / 12.f);
    out0[n * out0_stride + y * HW + x] =
        rintf(spf_clampf(q0 * 255.f, 0.f, 255.f));
    if constexpr (OC == 2) {
      const float q1 = accv[1] * (1.f / 12.f);
      out1[n * out1_stride + y * HW + x] =
          rintf(spf_clampf(q1 * 255.f, 0.f, 255.f));
    }
  }
}

// ---------------------------------------------------------------------------
// Stage 6: per LR pixel, sum over (m,c,r) of 16-channel LUT outputs, pixel-
// shuffled into the 4x4 HR block with the inverse rotation permutation.
// Tables (2.1 MB total) are read through L2 directly; A operands from LDS.
// ---------------------------------------------------------------------------
__global__ void __launch_bounds__(256)
spf_stage6_kernel(const float* __restrict__ feat,  // (2,4,192,192)
                  const float* __restrict__ sw,    // (3,4,4,3,3)
                  const float* __restrict__ w1,    // (3,4,2125,16)
                  const float* __restrict__ w2,    // (3,4,625,16)
                  const float* __restrict__ res,   // (3,4,16)
                  float* __restrict__ out) {       // (2,768,768)
  __shared__ v16h Als[12 * 32];     // (m*4+c) combos x 32 lanes

  fill_A_lds((_Float16*)Als, sw, 12);
  __syncthreads();

  const int lane = (int)threadIdx.x & 31;
  const int hi   = lane >> 4;
  const int wave = (int)blockIdx.x * 8 + ((int)threadIdx.x >> 5);
  const int p    = wave * 16 + (lane & 15);
  const int n    = p / IMG_N;
  const int rem  = p - n * IMG_N;
  const int y    = rem / HW;
  const int x    = rem - y * HW;

  float acc16[16];
#pragma unroll
  for (int k = 0; k < 16; ++k) acc16[k] = 0.f;

  for (int cc = 0; cc < 4; ++cc) {
    const v16h B = build_B(feat + (n * 4 + cc) * IMG_N, y, x, hi);
    for (int m = 0; m < 3; ++m) {
      const int combo = m * 4 + cc;
      const v16h A = Als[combo * 32 + lane];

      v8f C = {};
      C = __builtin_amdgcn_wmma_f32_16x16x32_f16(false, A, false, B, (short)0, C,
                                                 false, false);

      const float* w1g = w1 + combo * 2125 * 16;
      const float* w2g = w2 + combo * 625 * 16;
      const float* rm  = res + combo * 16;
#pragma unroll
      for (int g = 0; g < 2; ++g) {
        const int r = hi * 2 + g;
        float v4[4];
#pragma unroll
        for (int c = 0; c < 4; ++c) v4[c] = spf_clampf(C[g * 4 + c], 0.f, 255.f);
        float term[16];
        lut_term<16>(v4, w1g, w2g, term);
#pragma unroll
        for (int u = 0; u < 4; ++u) {
#pragma unroll
          for (int v = 0; v < 4; ++v) {
            const float t = term[u * 4 + v] + rm[u * 4 + v];
            int pp, qq;
            if (r == 0)      { pp = u;     qq = v;     }
            else if (r == 1) { pp = v;     qq = 3 - u; }
            else if (r == 2) { pp = 3 - u; qq = 3 - v; }
            else             { pp = 3 - v; qq = u;     }
            acc16[pp * 4 + qq] += t;
          }
        }
      }
    }
  }

#pragma unroll
  for (int k = 0; k < 16; ++k) acc16[k] += __shfl_xor(acc16[k], 16, 32);

  if (lane < 16) {
    float* ob = out + n * (768 * 768);
    const int Y0 = y * 4, X0 = x * 4;
#pragma unroll
    for (int pp = 0; pp < 4; ++pp) {
#pragma unroll
      for (int qq = 0; qq < 4; ++qq) {
        const float hr = acc16[pp * 4 + qq] * (1.f / 48.f);
        ob[(Y0 + pp) * 768 + X0 + qq] =
            spf_clampf(hr * 255.f, 0.f, 255.f) * (1.f / 255.f);
      }
    }
  }
}

__global__ void spf_quant_kernel(const float* __restrict__ x,
                                 float* __restrict__ cur, int nelem) {
  const int i = (int)blockIdx.x * (int)blockDim.x + (int)threadIdx.x;
  if (i < nelem) cur[i] = rintf(spf_clampf(x[i] * 255.f, 0.f, 255.f));
}

// ---------------------------------------------------------------------------
extern "C" void kernel_launch(void* const* d_in, const int* in_sizes, int n_in,
                              void* d_out, int out_size, void* d_ws,
                              size_t ws_size, hipStream_t stream) {
  (void)in_sizes; (void)n_in; (void)out_size; (void)ws_size;
  const float* x       = (const float*)d_in[0];
  // d_in[1] = ref2index: arange(R1) -> identity, computed arithmetically.
  const float* samp123 = (const float*)d_in[2];
  const float* w1123   = (const float*)d_in[3];
  const float* w2123   = (const float*)d_in[4];
  const float* res123  = (const float*)d_in[5];
  const float* samp4   = (const float*)d_in[6];
  const float* w14     = (const float*)d_in[7];
  const float* w24     = (const float*)d_in[8];
  const float* res4    = (const float*)d_in[9];
  const float* samp6   = (const float*)d_in[10];
  const float* w16     = (const float*)d_in[11];
  const float* w26     = (const float*)d_in[12];
  const float* res6    = (const float*)d_in[13];
  float* out = (float*)d_out;

  float* curA = (float*)d_ws;          // (2,192,192)
  float* curB = curA + NPIX;           // (2,192,192)
  float* feat = curB + NPIX;           // (2,4,192,192)

  spf_quant_kernel<<<(NPIX + 255) / 256, 256, 0, stream>>>(x, curA, NPIX);

  const int blocks = NPIX / 128;  // 8 waves/block * 16 px/wave = 128 px/block
  // stages 1..3 (OC=2): ch0 -> feat[:,s], ch1 -> next cur (ping-pong)
  spf_stage_kernel<2><<<blocks, 256, 0, stream>>>(
      curA, samp123 + 0 * 108, w1123 + 0 * 12750, w2123 + 0 * 3750,
      res123 + 0 * 6, feat + 0 * IMG_N, 4 * IMG_N, curB, IMG_N);
  spf_stage_kernel<2><<<blocks, 256, 0, stream>>>(
      curB, samp123 + 1 * 108, w1123 + 1 * 12750, w2123 + 1 * 3750,
      res123 + 1 * 6, feat + 1 * IMG_N, 4 * IMG_N, curA, IMG_N);
  spf_stage_kernel<2><<<blocks, 256, 0, stream>>>(
      curA, samp123 + 2 * 108, w1123 + 2 * 12750, w2123 + 2 * 3750,
      res123 + 2 * 6, feat + 2 * IMG_N, 4 * IMG_N, curB, IMG_N);
  // stage 4 (OC=1): -> feat[:,3]
  spf_stage_kernel<1><<<blocks, 256, 0, stream>>>(
      curB, samp4, w14, w24, res4, feat + 3 * IMG_N, 4 * IMG_N, nullptr, 0);
  // stage 6: 4x upsample -> d_out
  spf_stage6_kernel<<<blocks, 256, 0, stream>>>(feat, samp6, w16, w26, res6, out);
}
